// MultiHeadAttention_82446192214635
// MI455X (gfx1250) — matrix operand order repaired and checked by reading, not compiled, not run on hardware
//
#include <hip/hip_runtime.h>
#include <hip/hip_bf16.h>

// ---------------------------------------------------------------------------
// MHA forward for B=4, S=2048, E=1024, H=16, DH=64 on gfx1250 (CDNA5).
// All matmuls run on v_wmma_f32_16x16x32_bf16 (f32 accumulate).
// B-side operands (weights / K / V tiles) are block-shared: staged into LDS
// once per block (double-buffered, 1 barrier per k-step) via gfx1250
// global_load_async_to_lds_b128 (ASYNCcnt) when available.
// ---------------------------------------------------------------------------

#define Bb 4
#define Ss 2048
#define Ee 1024
#define Hh 16
#define Dh 64
#define HD (Hh * Dh)   // 1024

typedef __bf16 bf16_t;
typedef __attribute__((ext_vector_type(16))) __bf16 v16bf;
typedef __attribute__((ext_vector_type(8)))  __bf16 v8bf;
typedef __attribute__((ext_vector_type(8)))  float  v8f;
typedef __attribute__((ext_vector_type(4)))  int    v4i;

#if defined(__has_builtin)
#if __has_builtin(__builtin_amdgcn_global_load_async_to_lds_b128) && \
    __has_builtin(__builtin_amdgcn_s_wait_asynccnt)
#define USE_ASYNC_LDS 1
#endif
#endif
#ifndef USE_ASYNC_LDS
#define USE_ASYNC_LDS 0
#endif

// copy 16B global -> LDS (per lane)
static __device__ __forceinline__ void cp16_g2l(const bf16_t* g, bf16_t* l) {
#if USE_ASYNC_LDS
  __builtin_amdgcn_global_load_async_to_lds_b128(
      (__attribute__((address_space(1))) v4i*)(void*)const_cast<bf16_t*>(g),
      (__attribute__((address_space(3))) v4i*)(void*)l, 0, 0);
#else
  *(v8bf*)l = *(const v8bf*)g;
#endif
}

// make staged LDS data visible before the publishing barrier
static __device__ __forceinline__ void stage_fence() {
#if USE_ASYNC_LDS
  __builtin_amdgcn_s_wait_asynccnt(0);
#endif
}

static __device__ __forceinline__ bf16_t f2bf(float f) {
  union { float f; unsigned int u; } x; x.f = f;
  unsigned int r = x.u + 0x7FFFu + ((x.u >> 16) & 1u);  // round-to-nearest-even
  union { unsigned short s; bf16_t b; } y;
  y.s = (unsigned short)(r >> 16);
  return y.b;
}

// A-fragment (16x32 bf16, MxK): lane(0-15)=row M, K = {0..7,16..23};
// lane(16-31)=row M, K = {8..15,24..31}.  src row-major, ld in elements.
static __device__ __forceinline__ v16bf load_a_frag(const bf16_t* base, int ld) {
  const int lane = threadIdx.x & 31;
  const int row  = lane & 15;
  const int k0   = (lane >> 4) << 3;           // 0 or 8
  const bf16_t* p = base + (long)row * ld + k0;
  v8bf lo = *(const v8bf*)(p);                 // K = k0 .. k0+7
  v8bf hi = *(const v8bf*)(p + 16);            // K = k0+16 .. k0+23
  v16bf r;
#pragma unroll
  for (int i = 0; i < 8; ++i) { r[i] = lo[i]; r[i + 8] = hi[i]; }
  return r;
}

// B-fragment (32x16 bf16, KxN) loaded from N-major storage (row = output col n,
// contiguous along K, stride ld): lanes 0-15 hold K={0..7,16..23}, lanes 16-31 K={8..15,24..31}.
static __device__ __forceinline__ v16bf load_b_frag(const bf16_t* base, int ld) {
  const int lane = threadIdx.x & 31;
  const int col  = lane & 15;
  const int k0   = (lane >> 4) << 3;           // 0 or 8
  const bf16_t* p = base + (long)col * ld + k0;
  v8bf lo = *(const v8bf*)(p);                 // K = k0 .. k0+7
  v8bf hi = *(const v8bf*)(p + 16);            // K = k0+16 .. k0+23
  v16bf r;
#pragma unroll
  for (int i = 0; i < 8; ++i) { r[i] = lo[i]; r[i + 8] = hi[i]; }
  return r;
}

static __device__ __forceinline__ v8f wmma_bf16(v16bf a, v16bf b, v8f c) {
  return __builtin_amdgcn_wmma_f32_16x16x32_bf16(false, a, false, b,
                                                 (short)0, c, false, false);
}

// ---------------------------------------------------------------------------
// Stage 0: f32 -> bf16 conversion (grid-stride)
// ---------------------------------------------------------------------------
__global__ void k_cvt_bf16(const float* __restrict__ src,
                           bf16_t* __restrict__ dst, long n) {
  long i = (long)blockIdx.x * blockDim.x + threadIdx.x;
  const long stride = (long)gridDim.x * blockDim.x;
  for (; i < n; i += stride) dst[i] = f2bf(src[i]);
}

// dst[b][c][r] = bf16(src[b][r][c])   (batched transpose + convert)
__global__ void k_transpose_bf16(const float* __restrict__ src,
                                 bf16_t* __restrict__ dst,
                                 int rows, int cols) {
  const int b = blockIdx.y;
  const long n = (long)rows * cols;
  const float* s = src + (long)b * n;
  bf16_t* d = dst + (long)b * n;
  long i = (long)blockIdx.x * blockDim.x + threadIdx.x;
  const long stride = (long)gridDim.x * blockDim.x;
  for (; i < n; i += stride) {
    long c = i / rows, r = i - c * rows;       // dst flat index = c*rows + r
    d[i] = f2bf(s[r * (long)cols + c]);
  }
}

#define WROW 40   // LDS row pitch for 32-element rows (80B, 16B multiple)

// ---------------------------------------------------------------------------
// Stage 1: per-head projection  out = x @ W[h] + bias[h]
// ---------------------------------------------------------------------------
__global__ __launch_bounds__(128) void k_proj(const bf16_t* __restrict__ x,
                                              const bf16_t* __restrict__ wT,
                                              const float*  __restrict__ bias,
                                              bf16_t* __restrict__ out,
                                              int transposed) {
  __shared__ __align__(16) bf16_t wtile[2][64 * WROW];

  const int wave = threadIdx.x >> 5;
  const int lane = threadIdx.x & 31;
  const int tid  = threadIdx.x;
  const int h = blockIdx.y;
  const long row0 = (long)blockIdx.x * 128 + wave * 32;  // flat (b,s) row
  const int b  = (int)(row0 / Ss);
  const int s0 = (int)(row0 % Ss);

  const bf16_t* arow0 = x + row0 * Ee;
  const bf16_t* arow1 = arow0 + (long)16 * Ee;
  const bf16_t* wbase = wT + (long)h * Dh * Ee;

  v8f acc[2][4];
#pragma unroll
  for (int g = 0; g < 2; ++g)
#pragma unroll
    for (int t = 0; t < 4; ++t)
#pragma unroll
      for (int j = 0; j < 8; ++j) acc[g][t][j] = 0.f;

  // cooperative stage: 64 rows x 64B = 256 x 16B chunks, 2 per thread
  auto stage = [&](int kk, int buf) {
#pragma unroll
    for (int i = 0; i < 2; ++i) {
      const int c = tid * 2 + i;
      const int rowi = c >> 2;
      const int ch = c & 3;
      cp16_g2l(wbase + (long)rowi * Ee + kk + ch * 8,
               &wtile[buf][rowi * WROW + ch * 8]);
    }
  };

  auto body = [&](int k, int cur, v16bf& a0c, v16bf& a1c,
                  v16bf& a0n, v16bf& a1n, bool more) {
    if (more) {
      stage(k + 32, cur ^ 1);
      a0n = load_a_frag(arow0 + k + 32, Ee);
      a1n = load_a_frag(arow1 + k + 32, Ee);
    }
#pragma unroll
    for (int t = 0; t < 4; ++t) {
      v16bf bfr = load_b_frag(&wtile[cur][(t * 16) * WROW], WROW);
      acc[0][t] = wmma_bf16(a0c, bfr, acc[0][t]);
      acc[1][t] = wmma_bf16(a1c, bfr, acc[1][t]);
    }
    stage_fence();
    __syncthreads();
  };

  stage(0, 0);
  v16bf a0A = load_a_frag(arow0, Ee);
  v16bf a1A = load_a_frag(arow1, Ee);
  v16bf a0B, a1B;
  stage_fence();
  __syncthreads();

  int k = 0;
  for (; k < Ee - 64; k += 64) {
    __builtin_prefetch(arow0 + k + 96, 0, 3);
    __builtin_prefetch(arow1 + k + 96, 0, 3);
    body(k,      0, a0A, a1A, a0B, a1B, true);
    body(k + 32, 1, a0B, a1B, a0A, a1A, true);
  }
  body(k,      0, a0A, a1A, a0B, a1B, true);
  body(k + 32, 1, a0B, a1B, a0A, a1A, false);

  const int n  = lane & 15;
  const int mj = (lane >> 4) << 3;
#pragma unroll
  for (int g = 0; g < 2; ++g) {
    const int sg = s0 + g * 16;
#pragma unroll
    for (int t = 0; t < 4; ++t) {
      const int d = t * 16 + n;
      const float bv = bias[h * Dh + d];
#pragma unroll
      for (int j = 0; j < 8; ++j) {
        const int srow = sg + mj + j;
        const float v = acc[g][t][j] + bv;
        if (!transposed)
          out[(((long)b * Hh + h) * Ss + srow) * Dh + d] = f2bf(v);
        else
          out[(((long)b * Hh + h) * Dh + d) * Ss + srow] = f2bf(v);
      }
    }
  }
}

// ---------------------------------------------------------------------------
// Stage 2: flash attention per (b,h).  K/V tiles block-shared in LDS
// (double-buffered, 1 barrier/tile); 4 waves x 16 query rows, online softmax.
//   qh,kh : (B,H,S,DH) bf16 ; vt : (B,H,DH,S) bf16 ; abias : (S,S) f32
//   ctx   : (B,S,H*DH) bf16
// ---------------------------------------------------------------------------
#define KROW 72   // K-tile LDS pitch: 64-elem rows (144B, 16B multiple)

__global__ __launch_bounds__(128) void k_attn(const bf16_t* __restrict__ qh,
                                              const bf16_t* __restrict__ kh,
                                              const bf16_t* __restrict__ vt,
                                              const float*  __restrict__ abias,
                                              bf16_t* __restrict__ ctx) {
  __shared__ __align__(16) bf16_t kts[2][32 * KROW];   // 32 keys x 64 feat
  __shared__ __align__(16) bf16_t vts[2][64 * WROW];   // 64 d x 32 keys
  __shared__ __align__(16) bf16_t lds_p[4][16 * 32];   // per-wave P tile

  const int wave = threadIdx.x >> 5;
  const int lane = threadIdx.x & 31;
  const int tid  = threadIdx.x;
  const int bh = blockIdx.y;                 // b*H + h
  const int b = bh >> 4, h = bh & 15;
  const int q0 = blockIdx.x * 64 + wave * 16;

  const bf16_t* qbase = qh + ((long)bh * Ss + q0) * Dh;
  const bf16_t* kbase = kh + (long)bh * Ss * Dh;
  const bf16_t* vbase = vt + (long)bh * Dh * Ss;

  const v16bf aq0 = load_a_frag(qbase, Dh);        // k = 0..31
  const v16bf aq1 = load_a_frag(qbase + 32, Dh);   // k = 32..63

  v8f o[4];
  float mrow[8], lrow[8];
#pragma unroll
  for (int t = 0; t < 4; ++t)
#pragma unroll
    for (int j = 0; j < 8; ++j) o[t][j] = 0.f;
#pragma unroll
  for (int j = 0; j < 8; ++j) { mrow[j] = -3.0e38f; lrow[j] = 0.f; }

  const int n  = lane & 15;
  const int mj = (lane >> 4) << 3;
  const float scale = 0.125f;                // 1/sqrt(DH)
  bf16_t* pt = lds_p[wave];

  // cooperative stage of one 32-key K tile + V tile (256+256 16B chunks)
  auto stage_kv = [&](int jt, int buf) {
#pragma unroll
    for (int i = 0; i < 2; ++i) {            // K: 32 rows x 8 chunks
      const int c = tid * 2 + i;
      const int rowi = c >> 3;
      const int ch = c & 7;
      cp16_g2l(kbase + (long)(jt + rowi) * Dh + ch * 8,
               &kts[buf][rowi * KROW + ch * 8]);
    }
#pragma unroll
    for (int i = 0; i < 2; ++i) {            // V: 64 rows x 4 chunks
      const int c = tid * 2 + i;
      const int rowi = c >> 2;
      const int ch = c & 3;
      cp16_g2l(vbase + (long)rowi * Ss + jt + ch * 8,
               &vts[buf][rowi * WROW + ch * 8]);
    }
  };

  stage_kv(0, 0);
  stage_fence();
  __syncthreads();

  for (int it = 0; it < Ss / 32; ++it) {
    const int jt = it * 32;
    const int cur = it & 1;
    if (jt + 32 < Ss) stage_kv(jt + 32, cur ^ 1);   // fill other buffer

    // bias loads: in flight while the score WMMAs run
    float bia0[8], bia1[8];
#pragma unroll
    for (int j = 0; j < 8; ++j) {
      const float* brow = abias + (long)(q0 + mj + j) * Ss + jt;
      bia0[j] = brow[n];
      bia1[j] = brow[16 + n];
    }

    v8f s0, s1;
#pragma unroll
    for (int j = 0; j < 8; ++j) { s0[j] = 0.f; s1[j] = 0.f; }
    s0 = wmma_bf16(aq0, load_b_frag(&kts[cur][0], KROW), s0);
    s0 = wmma_bf16(aq1, load_b_frag(&kts[cur][32], KROW), s0);
    s1 = wmma_bf16(aq0, load_b_frag(&kts[cur][16 * KROW], KROW), s1);
    s1 = wmma_bf16(aq1, load_b_frag(&kts[cur][16 * KROW + 32], KROW), s1);

#pragma unroll
    for (int j = 0; j < 8; ++j) {
      s0[j] = s0[j] * scale + bia0[j];
      s1[j] = s1[j] * scale + bia1[j];
    }

    // online softmax update (per-row reductions across the 16-lane group)
#pragma unroll
    for (int j = 0; j < 8; ++j) {
      float tm = fmaxf(s0[j], s1[j]);
      tm = fmaxf(tm, __shfl_xor(tm, 1, 32));
      tm = fmaxf(tm, __shfl_xor(tm, 2, 32));
      tm = fmaxf(tm, __shfl_xor(tm, 4, 32));
      tm = fmaxf(tm, __shfl_xor(tm, 8, 32));
      const float mnew = fmaxf(mrow[j], tm);
      const float corr = __expf(mrow[j] - mnew);
      mrow[j] = mnew;
      const float p0 = __expf(s0[j] - mnew);
      const float p1 = __expf(s1[j] - mnew);
      float ps = p0 + p1;
      ps += __shfl_xor(ps, 1, 32);
      ps += __shfl_xor(ps, 2, 32);
      ps += __shfl_xor(ps, 4, 32);
      ps += __shfl_xor(ps, 8, 32);
      lrow[j] = lrow[j] * corr + ps;
#pragma unroll
      for (int t = 0; t < 4; ++t) o[t][j] *= corr;
      pt[(mj + j) * 32 + n]      = f2bf(p0);
      pt[(mj + j) * 32 + 16 + n] = f2bf(p1);
    }

    // P (16x32) back as an A-fragment (same-wave LDS ops are in-order)
    v16bf ap = load_a_frag(pt, 32);
#pragma unroll
    for (int t = 0; t < 4; ++t)
      o[t] = wmma_bf16(ap, load_b_frag(&vts[cur][(t * 16) * WROW], WROW), o[t]);

    stage_fence();
    __syncthreads();   // cur fully consumed; next buffer fully staged
  }

  // normalize + store into concat layout (B,S,H*DH)
#pragma unroll
  for (int j = 0; j < 8; ++j) {
    const float inv = 1.0f / lrow[j];
    const long orow = ((long)b * Ss + q0 + mj + j) * HD + h * Dh;
#pragma unroll
    for (int t = 0; t < 4; ++t)
      ctx[orow + t * 16 + n] = f2bf(o[t][j] * inv);
  }
}

// ---------------------------------------------------------------------------
// Stage 3: output projection  out = ctx @ Wo + bo   (f32 output)
// ---------------------------------------------------------------------------
__global__ __launch_bounds__(128) void k_outproj(const bf16_t* __restrict__ ctx,
                                                 const bf16_t* __restrict__ woT,
                                                 const float* __restrict__ bo,
                                                 float* __restrict__ out) {
  __shared__ __align__(16) bf16_t wtile[2][64 * WROW];

  const int wave = threadIdx.x >> 5;
  const int lane = threadIdx.x & 31;
  const int tid  = threadIdx.x;
  const long row0 = (long)blockIdx.x * 128 + wave * 32;
  const int e0 = blockIdx.y * 64;
  const bf16_t* arow0 = ctx + row0 * HD;
  const bf16_t* arow1 = arow0 + (long)16 * HD;
  const bf16_t* wbase = woT + (long)e0 * HD;

  v8f acc[2][4];
#pragma unroll
  for (int g = 0; g < 2; ++g)
#pragma unroll
    for (int t = 0; t < 4; ++t)
#pragma unroll
      for (int j = 0; j < 8; ++j) acc[g][t][j] = 0.f;

  auto stage = [&](int kk, int buf) {
#pragma unroll
    for (int i = 0; i < 2; ++i) {
      const int c = tid * 2 + i;
      const int rowi = c >> 2;
      const int ch = c & 3;
      cp16_g2l(wbase + (long)rowi * HD + kk + ch * 8,
               &wtile[buf][rowi * WROW + ch * 8]);
    }
  };

  auto body = [&](int k, int cur, v16bf& a0c, v16bf& a1c,
                  v16bf& a0n, v16bf& a1n, bool more) {
    if (more) {
      stage(k + 32, cur ^ 1);
      a0n = load_a_frag(arow0 + k + 32, HD);
      a1n = load_a_frag(arow1 + k + 32, HD);
    }
#pragma unroll
    for (int t = 0; t < 4; ++t) {
      v16bf bfr = load_b_frag(&wtile[cur][(t * 16) * WROW], WROW);
      acc[0][t] = wmma_bf16(a0c, bfr, acc[0][t]);
      acc[1][t] = wmma_bf16(a1c, bfr, acc[1][t]);
    }
    stage_fence();
    __syncthreads();
  };

  stage(0, 0);
  v16bf a0A = load_a_frag(arow0, HD);
  v16bf a1A = load_a_frag(arow1, HD);
  v16bf a0B, a1B;
  stage_fence();
  __syncthreads();

  int k = 0;
  for (; k < HD - 64; k += 64) {
    __builtin_prefetch(arow0 + k + 96, 0, 3);
    __builtin_prefetch(arow1 + k + 96, 0, 3);
    body(k,      0, a0A, a1A, a0B, a1B, true);
    body(k + 32, 1, a0B, a1B, a0A, a1A, true);
  }
  body(k,      0, a0A, a1A, a0B, a1B, true);
  body(k + 32, 1, a0B, a1B, a0A, a1A, false);

  const int n  = lane & 15;
  const int mj = (lane >> 4) << 3;
#pragma unroll
  for (int g = 0; g < 2; ++g) {
#pragma unroll
    for (int t = 0; t < 4; ++t) {
      const float bv = bo[e0 + t * 16 + n];
#pragma unroll
      for (int j = 0; j < 8; ++j)
        out[(row0 + g * 16 + mj + j) * Ee + e0 + t * 16 + n] = acc[g][t][j] + bv;
    }
  }
}

// ---------------------------------------------------------------------------
extern "C" void kernel_launch(void* const* d_in, const int* in_sizes, int n_in,
                              void* d_out, int out_size, void* d_ws, size_t ws_size,
                              hipStream_t stream) {
  const float* q_f  = (const float*)d_in[0];
  const float* k_f  = (const float*)d_in[1];
  const float* v_f  = (const float*)d_in[2];
  const float* ab   = (const float*)d_in[3];
  const float* Wq   = (const float*)d_in[4];
  const float* bq   = (const float*)d_in[5];
  const float* Wk   = (const float*)d_in[6];
  const float* bk   = (const float*)d_in[7];
  const float* Wv   = (const float*)d_in[8];
  const float* bv   = (const float*)d_in[9];
  const float* Wo   = (const float*)d_in[10];
  const float* bo   = (const float*)d_in[11];
  float* out = (float*)d_out;

  const long nBSE = (long)Bb * Ss * Ee;        // 8,388,608
  const long nW   = (long)Hh * Ee * Dh;        // 1,048,576

  size_t off = 0;
  auto wsalloc = [&](size_t bytes) {
    void* p = (char*)d_ws + off;
    off += (bytes + 255) & ~(size_t)255;
    return p;
  };
  bf16_t* qbf  = (bf16_t*)wsalloc(nBSE * 2);
  bf16_t* kbf  = (bf16_t*)wsalloc(nBSE * 2);
  bf16_t* vbf  = (bf16_t*)wsalloc(nBSE * 2);
  bf16_t* wqT  = (bf16_t*)wsalloc(nW * 2);
  bf16_t* wkT  = (bf16_t*)wsalloc(nW * 2);
  bf16_t* wvT  = (bf16_t*)wsalloc(nW * 2);
  bf16_t* woT  = (bf16_t*)wsalloc((long)HD * Ee * 2);
  bf16_t* qhd  = (bf16_t*)wsalloc(nBSE * 2);   // (B,H,S,DH)
  bf16_t* khd  = (bf16_t*)wsalloc(nBSE * 2);   // (B,H,S,DH)
  bf16_t* vtd  = (bf16_t*)wsalloc(nBSE * 2);   // (B,H,DH,S)
  bf16_t* ctxb = (bf16_t*)wsalloc(nBSE * 2);   // (B,S,H*DH)
  (void)ws_size; (void)in_sizes; (void)n_in; (void)out_size;

  // Stage 0: conversions / weight transposes
  k_cvt_bf16<<<1024, 256, 0, stream>>>(q_f, qbf, nBSE);
  k_cvt_bf16<<<1024, 256, 0, stream>>>(k_f, kbf, nBSE);
  k_cvt_bf16<<<1024, 256, 0, stream>>>(v_f, vbf, nBSE);
  k_transpose_bf16<<<dim3(256, Hh), 256, 0, stream>>>(Wq, wqT, Ee, Dh);
  k_transpose_bf16<<<dim3(256, Hh), 256, 0, stream>>>(Wk, wkT, Ee, Dh);
  k_transpose_bf16<<<dim3(256, Hh), 256, 0, stream>>>(Wv, wvT, Ee, Dh);
  k_transpose_bf16<<<dim3(256, 1), 256, 0, stream>>>(Wo, woT, HD, Ee);

  // Stage 1: Q/K/V projections (grid: 8192/128 row-tiles x H heads)
  k_proj<<<dim3(64, Hh), 128, 0, stream>>>(qbf, wqT, bq, qhd, 0);
  k_proj<<<dim3(64, Hh), 128, 0, stream>>>(kbf, wkT, bk, khd, 0);
  k_proj<<<dim3(64, Hh), 128, 0, stream>>>(vbf, wvT, bv, vtd, 1);

  // Stage 2: flash attention (grid: S/64 query-tiles x B*H)
  k_attn<<<dim3(Ss / 64, Bb * Hh), 128, 0, stream>>>(qhd, khd, vtd, ab, ctxb);

  // Stage 3: output projection (grid: 8192/128 row-tiles x 1024/64 col-tiles)
  k_outproj<<<dim3((Bb * Ss) / 128, Ee / 64), 128, 0, stream>>>(ctxb, woT, bo, out);
}
